// Texture_Encoder_39651138077473
// MI455X (gfx1250) — matrix operand order repaired and checked by reading, not compiled, not run on hardware
//
#include <hip/hip_runtime.h>
#include <hip/hip_bf16.h>
#include <math.h>

// ---------------------------------------------------------------------------
// Texture encoder, Gram-matrix formulation:
//   sm_i = softmax(10*w1_i, axis=C)                 (softmax_kernel)
//   G_i[n] = M M^T, M = x_i[n] reshaped (C, HW)     (gram_kernel, bf16 WMMA,
//                                                    64x64 reg tile, K-split)
//   feat_i[n,o] = sm[:,o]^T G_i[n] sm[:,o]          (quadfeat_kernel)
//   h = concat(norm(feat_i)); 3-layer MLP w/ norms  (mlp_kernel)
// ---------------------------------------------------------------------------

typedef __attribute__((ext_vector_type(16))) __bf16 v16bf;
typedef __attribute__((ext_vector_type(8)))  float  v8f;

#define NQ   128
#define NB   8      // batch
#define DIN  640    // 5*NQ

// Per-scale partial-G region size in floats: nsplit * NB * C * C == 2M for all.
#define GP_STRIDE 2097152u

// ---------------- softmax over channel axis (per output column) -------------
__global__ __launch_bounds__(128) void softmax_kernel(
    const float* __restrict__ w, float* __restrict__ sm, int C)
{
    const int o   = blockIdx.x;      // 0..127
    const int tid = threadIdx.x;     // 128 threads
    __shared__ float red[128];

    float mx = -INFINITY;
    for (int c = tid; c < C; c += 128) mx = fmaxf(mx, w[c * NQ + o] * 10.0f);
    red[tid] = mx; __syncthreads();
    for (int s = 64; s > 0; s >>= 1) {
        if (tid < s) red[tid] = fmaxf(red[tid], red[tid + s]);
        __syncthreads();
    }
    mx = red[0]; __syncthreads();

    float sum = 0.0f;
    for (int c = tid; c < C; c += 128) sum += expf(w[c * NQ + o] * 10.0f - mx);
    red[tid] = sum; __syncthreads();
    for (int s = 64; s > 0; s >>= 1) {
        if (tid < s) red[tid] += red[tid + s];
        __syncthreads();
    }
    const float inv = 1.0f / red[0];

    for (int c = tid; c < C; c += 128)
        sm[c * NQ + o] = expf(w[c * NQ + o] * 10.0f - mx) * inv;
}

// ---------------- fp32x16 -> bf16x16 operand pack ---------------------------
__device__ __forceinline__ v16bf pack_bf16(float4 f0, float4 f1, float4 f2, float4 f3)
{
    v16bf r;
    r[0]  = (__bf16)f0.x; r[1]  = (__bf16)f0.y; r[2]  = (__bf16)f0.z; r[3]  = (__bf16)f0.w;
    r[4]  = (__bf16)f1.x; r[5]  = (__bf16)f1.y; r[6]  = (__bf16)f1.z; r[7]  = (__bf16)f1.w;
    r[8]  = (__bf16)f2.x; r[9]  = (__bf16)f2.y; r[10] = (__bf16)f2.z; r[11] = (__bf16)f2.w;
    r[12] = (__bf16)f3.x; r[13] = (__bf16)f3.y; r[14] = (__bf16)f3.z; r[15] = (__bf16)f3.w;
    return r;
}

// ---------------- Gram matrices via bf16 WMMA --------------------------------
// One merged launch for all 5 scales: 512 waves per scale, 2560 waves total.
// Each wave computes a 64x64 partial-G tile (4x4 grid of 16x16 WMMA tiles)
// over a K-range of HW/nsplit, writing to its own split's partial buffer.
//
// Operand VGPR layouts (ISA 7.12.2, 16-bit, wave32):
//   A (16x32): lane L holds row ci+(L&15), K = {ka..ka+7, 16+ka..16+ka+7}, ka=L<16?0:8
//   B (32x16): lane L holds col cj+(L&15) = row of A^T, K = {kb..kb+7, 16+kb..16+kb+7}, kb=L<16?0:8
__global__ __launch_bounds__(256) void gram_kernel(
    const float* __restrict__ x0, const float* __restrict__ x1,
    const float* __restrict__ x2, const float* __restrict__ x3,
    const float* __restrict__ x4, float* __restrict__ gp)
{
    const int wave = blockIdx.x * 8 + (threadIdx.x >> 5);
    const int lane = threadIdx.x & 31;

    const int s  = wave >> 9;      // 512 waves per scale
    const int id = wave & 511;

    // compile-time per-scale tables (selected by wave-uniform s)
    const int C   = (s == 0) ? 64    : (s == 1) ? 128   : (s == 2) ? 256  : 512;
    const int HW  = (s == 0) ? 65536 : (s == 1) ? 16384 : (s == 2) ? 4096 :
                    (s == 3) ? 1024  : 256;
    const int T64 = (s == 0) ? 1 : (s == 1) ? 2 : (s == 2) ? 4 : 8;
    const int KL  = (s == 4) ? 256 : 1024;            // HW / nsplit
    const float* __restrict__ xp =
        (s == 0) ? x0 : (s == 1) ? x1 : (s == 2) ? x2 : (s == 3) ? x3 : x4;

    const int n    = id >> 6;                 // 64 work items per sample
    const int rem  = id & 63;
    const int t2   = T64 * T64;
    const int split = rem / t2;
    const int tidx  = rem - split * t2;
    const int ti = tidx / T64;
    const int tj = tidx - ti * T64;

    const float* __restrict__ xa =
        xp + (size_t)n * C * HW + (size_t)(ti * 64 + (lane & 15)) * HW;
    const float* __restrict__ xb =
        xp + (size_t)n * C * HW + (size_t)(tj * 64 + (lane & 15)) * HW;
    const int ka = (lane < 16) ? 0 : 8;
    const int kb = (lane < 16) ? 0 : 8;
    const int kbeg = split * KL;
    const int kend = kbeg + KL;

    v8f acc[4][4];
    #pragma unroll
    for (int bi = 0; bi < 4; ++bi)
        #pragma unroll
        for (int bj = 0; bj < 4; ++bj) acc[bi][bj] = (v8f){};

    for (int k = kbeg; k < kend; k += 32) {
        __builtin_prefetch(xa + k + 128, 0, 3);   // next chunks toward WGP
        __builtin_prefetch(xb + k + 128, 0, 3);

        v16bf A[4], B[4];
        #pragma unroll
        for (int bi = 0; bi < 4; ++bi) {
            const float* p = xa + (size_t)(bi * 16) * HW + k;
            A[bi] = pack_bf16(*(const float4*)(p + ka),
                              *(const float4*)(p + ka + 4),
                              *(const float4*)(p + 16 + ka),
                              *(const float4*)(p + 16 + ka + 4));
        }
        #pragma unroll
        for (int bj = 0; bj < 4; ++bj) {
            const float* p = xb + (size_t)(bj * 16) * HW + k;
            B[bj] = pack_bf16(*(const float4*)(p + kb),
                              *(const float4*)(p + kb + 4),
                              *(const float4*)(p + 16 + kb),
                              *(const float4*)(p + 16 + kb + 4));
        }
        #pragma unroll
        for (int bi = 0; bi < 4; ++bi)
            #pragma unroll
            for (int bj = 0; bj < 4; ++bj)
                acc[bi][bj] = __builtin_amdgcn_wmma_f32_16x16x32_bf16(
                    false, A[bi], false, B[bj],
                    (short)0, acc[bi][bj], false, false);
    }

    // D layout: lane l holds N=(l&15); VGPR v holds M = v + (l<16 ? 0 : 8)
    const int mofs = (lane < 16) ? 0 : 8;
    float* __restrict__ g =
        gp + (size_t)s * GP_STRIDE + ((size_t)split * NB + n) * (size_t)(C * C);
    #pragma unroll
    for (int bi = 0; bi < 4; ++bi)
        #pragma unroll
        for (int bj = 0; bj < 4; ++bj) {
            float* gt = g + (size_t)(ti * 64 + bi * 16 + mofs) * C
                          + (tj * 64 + bj * 16 + (lane & 15));
            #pragma unroll
            for (int v = 0; v < 8; ++v) gt[(size_t)v * C] = acc[bi][bj][v];
        }
}

// ---------------- deterministic K-split reduction ---------------------------
// gp layout per scale: [nsplit][NB*C*C]; sum splits into split 0, fixed order.
__global__ __launch_bounds__(256) void reduce_gram(
    float* __restrict__ gp, int elems, int nsplit)
{
    const int i = blockIdx.x * 256 + threadIdx.x;
    if (i >= elems) return;
    float s = gp[i];
    for (int sp = 1; sp < nsplit; ++sp) s += gp[(size_t)sp * elems + i];
    gp[i] = s;
}

// ---------------- feat[n,o] = sm_o^T G[n] sm_o ------------------------------
__global__ __launch_bounds__(256) void quadfeat_kernel(
    const float* __restrict__ G, const float* __restrict__ sm,
    float* __restrict__ hrow /* hraw + scale*128 */, int C)
{
    const int n  = blockIdx.x;
    const int o0 = blockIdx.y * 16;
    const int tid = threadIdx.x;

    __shared__ float smc[512 * 16];      // sm[:, o0:o0+16], 32 KB max
    __shared__ float red[16][256];       // deterministic tree input

    for (int idx = tid; idx < C * 16; idx += 256) {
        const int c = idx >> 4, o = idx & 15;
        smc[idx] = sm[c * NQ + o0 + o];
    }
    __syncthreads();

    const float* __restrict__ Gn = G + (size_t)n * C * C;
    float part[16];
    #pragma unroll
    for (int o = 0; o < 16; ++o) part[o] = 0.0f;

    for (int c = tid; c < C; c += 256) {
        float t[16];
        #pragma unroll
        for (int o = 0; o < 16; ++o) t[o] = 0.0f;
        const float* __restrict__ grow = Gn + (size_t)c * C;
        for (int cp = 0; cp < C; ++cp) {
            const float gv = grow[cp];
            const float* s16 = &smc[cp * 16];
            #pragma unroll
            for (int o = 0; o < 16; ++o) t[o] = fmaf(gv, s16[o], t[o]);
        }
        #pragma unroll
        for (int o = 0; o < 16; ++o) part[o] = fmaf(smc[c * 16 + o], t[o], part[o]);
    }

    #pragma unroll
    for (int o = 0; o < 16; ++o) red[o][tid] = part[o];
    __syncthreads();
    if (tid < 16) {
        float acc = 0.0f;
        for (int j = 0; j < 256; ++j) acc += red[tid][j];
        hrow[n * DIN + o0 + tid] = acc;
    }
}

// ---------------- per-row norm helper ---------------------------------------
__device__ __forceinline__ float wave_sum32(float v) {
    #pragma unroll
    for (int off = 16; off > 0; off >>= 1) v += __shfl_xor(v, off, 32);
    return v;
}

// ---------------- feats-norm + MLP (tiny; one workgroup) --------------------
__global__ __launch_bounds__(256) void mlp_kernel(
    const float* __restrict__ hraw,
    const float* __restrict__ fc1w, const float* __restrict__ fc1b,
    const float* __restrict__ fc2w, const float* __restrict__ fc2b,
    const float* __restrict__ fc3w, const float* __restrict__ fc3b,
    float* __restrict__ out)
{
    __shared__ float h[NB * DIN];    // 20 KB
    __shared__ float a1[NB * 512];   // 16 KB
    __shared__ float a2[NB * 512];   // 16 KB
    const int tid  = threadIdx.x;
    const int wid  = tid >> 5;
    const int lane = tid & 31;

    for (int idx = tid; idx < NB * DIN; idx += 256) h[idx] = hraw[idx];
    __syncthreads();

    // normalize each (n, scale) chunk of 128 (ddof=1), one wave per row
    for (int r = wid; r < NB * 5; r += 8) {
        float* p = &h[(r / 5) * DIN + (r % 5) * NQ];
        float s = 0.0f;
        for (int j = lane; j < NQ; j += 32) s += p[j];
        const float mean = wave_sum32(s) / (float)NQ;
        float q = 0.0f;
        for (int j = lane; j < NQ; j += 32) { const float d = p[j] - mean; q += d * d; }
        const float sd = sqrtf(wave_sum32(q) / (float)(NQ - 1)) + 1e-8f;
        for (int j = lane; j < NQ; j += 32) p[j] = (p[j] - mean) / sd;
    }
    __syncthreads();

    // layer 1: (8x640) @ (640x512) + b
    for (int idx = tid; idx < NB * 512; idx += 256) {
        const int n = idx >> 9, j = idx & 511;
        float acc = fc1b[j];
        for (int i = 0; i < DIN; ++i) acc = fmaf(h[n * DIN + i], fc1w[i * 512 + j], acc);
        a1[idx] = acc;
    }
    __syncthreads();
    for (int r = wid; r < NB; r += 8) {      // norm over 512 + leaky(0.01)
        float* p = &a1[r * 512];
        float s = 0.0f;
        for (int j = lane; j < 512; j += 32) s += p[j];
        const float mean = wave_sum32(s) / 512.0f;
        float q = 0.0f;
        for (int j = lane; j < 512; j += 32) { const float d = p[j] - mean; q += d * d; }
        const float sd = sqrtf(wave_sum32(q) / 511.0f) + 1e-8f;
        for (int j = lane; j < 512; j += 32) {
            const float v = (p[j] - mean) / sd;
            p[j] = (v > 0.0f) ? v : 0.01f * v;
        }
    }
    __syncthreads();

    // layer 2: (8x512) @ (512x512) + b
    for (int idx = tid; idx < NB * 512; idx += 256) {
        const int n = idx >> 9, j = idx & 511;
        float acc = fc2b[j];
        for (int i = 0; i < 512; ++i) acc = fmaf(a1[n * 512 + i], fc2w[i * 512 + j], acc);
        a2[idx] = acc;
    }
    __syncthreads();
    for (int r = wid; r < NB; r += 8) {
        float* p = &a2[r * 512];
        float s = 0.0f;
        for (int j = lane; j < 512; j += 32) s += p[j];
        const float mean = wave_sum32(s) / 512.0f;
        float q = 0.0f;
        for (int j = lane; j < 512; j += 32) { const float d = p[j] - mean; q += d * d; }
        const float sd = sqrtf(wave_sum32(q) / 511.0f) + 1e-8f;
        for (int j = lane; j < 512; j += 32) {
            const float v = (p[j] - mean) / sd;
            p[j] = (v > 0.0f) ? v : 0.01f * v;
        }
    }
    __syncthreads();

    // layer 3: (8x512) @ (512x128) + b -> out
    for (int idx = tid; idx < NB * NQ; idx += 256) {
        const int n = idx >> 7, j = idx & 127;
        float acc = fc3b[j];
        for (int i = 0; i < 512; ++i) acc = fmaf(a2[n * 512 + i], fc3w[i * NQ + j], acc);
        out[idx] = acc;
    }
}

// ---------------------------------------------------------------------------
extern "C" void kernel_launch(void* const* d_in, const int* in_sizes, int n_in,
                              void* d_out, int out_size, void* d_ws, size_t ws_size,
                              hipStream_t stream) {
    (void)in_sizes; (void)n_in; (void)out_size; (void)ws_size;
    static const int CH[5] = {64, 128, 256, 512, 512};
    static const int NS[5] = {64, 16, 4, 1, 1};          // K-splits per scale

    float* ws = (float*)d_ws;
    size_t smOff[5], off = 0;
    for (int i = 0; i < 5; ++i) { smOff[i] = off; off += (size_t)CH[i] * NQ; }
    float* gp   = ws + off;  off += 5u * GP_STRIDE;      // partial/final G
    float* hraw = ws + off;                              // NB * DIN floats

    // dict order: x0,w1_0, x1,w1_1, ..., x4,w1_4, fc1_w,fc1_b, fc2_w,fc2_b, fc3_w,fc3_b
    for (int i = 0; i < 5; ++i) {
        const float* w1 = (const float*)d_in[2 * i + 1];
        softmax_kernel<<<NQ, 128, 0, stream>>>(w1, ws + smOff[i], CH[i]);
    }

    // merged Gram launch: 2560 waves = 320 blocks of 8 waves
    gram_kernel<<<320, 256, 0, stream>>>(
        (const float*)d_in[0], (const float*)d_in[2], (const float*)d_in[4],
        (const float*)d_in[6], (const float*)d_in[8], gp);

    // deterministic split reductions (scales with nsplit > 1)
    for (int i = 0; i < 5; ++i) {
        if (NS[i] > 1) {
            const int elems = NB * CH[i] * CH[i];
            reduce_gram<<<(elems + 255) / 256, 256, 0, stream>>>(
                gp + (size_t)i * GP_STRIDE, elems, NS[i]);
        }
    }

    for (int i = 0; i < 5; ++i) {
        dim3 grid(NB, NQ / 16);
        quadfeat_kernel<<<grid, 256, 0, stream>>>(
            gp + (size_t)i * GP_STRIDE, ws + smOff[i], hraw + i * NQ, CH[i]);
    }

    mlp_kernel<<<1, 256, 0, stream>>>(
        hraw,
        (const float*)d_in[10], (const float*)d_in[11],
        (const float*)d_in[12], (const float*)d_in[13],
        (const float*)d_in[14], (const float*)d_in[15],
        (float*)d_out);
}
